// BeepKcElectraResMLayer_46205258170733
// MI455X (gfx1250) — compile-verified, matrix operand order checked by reading, not run on hardware
//
#include <hip/hip_runtime.h>
#include <hip/hip_bf16.h>

// ---------------------------------------------------------------------------
// B=4 S=512 D=768 FF=3072 H=12 HD=64 | PKM: PH=4 KD=512 HALF=256 NK=128 KNN=32
// T = B*S = 2048
// All GEMMs run in bf16 via v_wmma_f32_16x16x32_bf16; weights are
// pre-transposed+converted once so every B operand is row-major N x K.
// Tiles are staged into LDS with global_load_async_to_lds_b128 (ASYNCcnt).
// ---------------------------------------------------------------------------

typedef __attribute__((ext_vector_type(16))) __bf16 v16bf;
typedef __attribute__((ext_vector_type(8)))  __bf16 v8bf;
typedef __attribute__((ext_vector_type(8)))  float  v8f;
typedef int v4i_ __attribute__((vector_size(16)));

#define BM 128
#define BN 128
#define BKT 32

#if defined(__gfx1250__) && __has_builtin(__builtin_amdgcn_global_load_async_to_lds_b128) && __has_builtin(__builtin_amdgcn_s_wait_asynccnt)
#define USE_ASYNC_LDS 1
#define AS1 __attribute__((address_space(1)))
#define AS3 __attribute__((address_space(3)))
#endif

// 16x32 bf16 fragment (CDNA5 A-layout) from LDS tile stored 32 bf16/row.
__device__ __forceinline__ v16bf load_frag(const __bf16* __restrict__ base,
                                           int row0, int lane) {
  int hi = (lane >> 4) & 1;
  int m  = row0 + (lane & 15);
  const __bf16* p = base + m * BKT + hi * 8;
  v8bf lo  = *reinterpret_cast<const v8bf*>(p);
  v8bf hi8 = *reinterpret_cast<const v8bf*>(p + 16);
  return __builtin_shufflevector(lo, hi8,
                                 0, 1, 2, 3, 4, 5, 6, 7,
                                 8, 9, 10, 11, 12, 13, 14, 15);
}

// Stage one 128x32 bf16 tile (rows from a row-major matrix, ld elements)
// into LDS. 256 threads x 2 chunks of 16B. Rows >= bound are zero-filled.
__device__ __forceinline__ void stage_tile(__bf16* __restrict__ lbase,
                                           const __bf16* __restrict__ gbase,
                                           long long ld, int r0, int bound,
                                           int k0, int tid) {
#pragma unroll
  for (int p = 0; p < 2; p++) {
    int c    = tid + p * 256;      // 512 chunks
    int row  = c >> 2;
    int q16  = c & 3;
    int grow = r0 + row;
    __bf16* lp = lbase + row * BKT + q16 * 8;
    if (grow < bound) {
      const __bf16* gp = gbase + (long long)grow * ld + k0 + q16 * 8;
#if USE_ASYNC_LDS
      __builtin_amdgcn_global_load_async_to_lds_b128(
          (AS1 v4i_*)gp, (AS3 v4i_*)lp, 0, 0);
#else
      *reinterpret_cast<float4*>(lp) = *reinterpret_cast<const float4*>(gp);
#endif
    } else {
      float4 z = make_float4(0.f, 0.f, 0.f, 0.f);
      *reinterpret_cast<float4*>(lp) = z;
    }
  }
}

// Batched bf16 GEMM: C = alpha * A(MxK) @ B(NxK)^T + bias, opt. exact GELU.
// off(z) = (z/zdiv)*outer + (z%zdiv)*inner for A/B/C.
// cmode: 0 = f32 store, 1 = bf16 store, 2 = bf16 store in (B,D,S) v-transpose.
__global__ __launch_bounds__(256)
void gemm_bf16_wmma_kernel(const __bf16* __restrict__ A, long long lda, long long aOut, long long aIn,
                           const __bf16* __restrict__ Bm, long long ldb, long long bOut, long long bIn,
                           void* __restrict__ Cv, long long ldc, long long cOut, long long cIn,
                           const float* __restrict__ bias,
                           int M, int N, int K, float alpha, int act, int zdiv, int cmode) {
  __shared__ __align__(16) __bf16 As[2][BM * BKT];
  __shared__ __align__(16) __bf16 Bs[2][BN * BKT];

  const int z = blockIdx.z;
  const __bf16* Ab = A  + (long long)(z / zdiv) * aOut + (long long)(z % zdiv) * aIn;
  const __bf16* Bb = Bm + (long long)(z / zdiv) * bOut + (long long)(z % zdiv) * bIn;

  const int m0  = blockIdx.y * BM;
  const int n0  = blockIdx.x * BN;
  const int tid = threadIdx.x;
  const int lane = tid & 31;
  const int wave = tid >> 5;
  const int wm = wave & 3;
  const int wn = wave >> 2;

  const v8f vzero = {0.f, 0.f, 0.f, 0.f, 0.f, 0.f, 0.f, 0.f};
  v8f acc[2][4];
#pragma unroll
  for (int i = 0; i < 2; i++)
#pragma unroll
    for (int j = 0; j < 4; j++) acc[i][j] = vzero;

  const int nk = K / BKT;
  stage_tile(&As[0][0], Ab, lda, m0, M, 0, tid);
  stage_tile(&Bs[0][0], Bb, ldb, n0, N, 0, tid);

  int cur = 0;
  for (int i = 0; i < nk; i++) {
#if USE_ASYNC_LDS
    __builtin_amdgcn_s_wait_asynccnt(0);
#endif
    __syncthreads();
    if (i + 1 < nk) {
      stage_tile(&As[cur ^ 1][0], Ab, lda, m0, M, (i + 1) * BKT, tid);
      stage_tile(&Bs[cur ^ 1][0], Bb, ldb, n0, N, (i + 1) * BKT, tid);
    }
    if (i + 2 < nk) {  // prefetch two tiles ahead into L2 (global_prefetch_b8)
      int rr = tid & 127;
      const __bf16* pf = (tid < 128)
          ? (Ab + (long long)(m0 + rr) * lda + (i + 2) * BKT)
          : (Bb + (long long)(n0 + rr) * ldb + (i + 2) * BKT);
      __builtin_prefetch((const void*)pf, 0, 1);
    }

    v16bf afr[2], bfr[4];
#pragma unroll
    for (int mt = 0; mt < 2; mt++) afr[mt] = load_frag(&As[cur][0], wm * 32 + mt * 16, lane);
#pragma unroll
    for (int nt = 0; nt < 4; nt++) bfr[nt] = load_frag(&Bs[cur][0], wn * 64 + nt * 16, lane);
#pragma unroll
    for (int mt = 0; mt < 2; mt++)
#pragma unroll
      for (int nt = 0; nt < 4; nt++)
        acc[mt][nt] = __builtin_amdgcn_wmma_f32_16x16x32_bf16(
            false, afr[mt], false, bfr[nt], (short)0, acc[mt][nt], false, false);
    cur ^= 1;
  }

  // epilogue
  float*  Cf = (float*) Cv + (long long)(z / zdiv) * cOut + (long long)(z % zdiv) * cIn;
  __bf16* Cb = (__bf16*)Cv + (long long)(z / zdiv) * cOut + (long long)(z % zdiv) * cIn;
  __bf16* Cvt = (__bf16*)Cv;  // cmode 2: (B, D, S) layout, batch z unused
  const int rowadd = ((lane >> 4) & 1) * 8;
  const int colin  = lane & 15;
#pragma unroll
  for (int mt = 0; mt < 2; mt++) {
#pragma unroll
    for (int nt = 0; nt < 4; nt++) {
#pragma unroll
      for (int r = 0; r < 8; r++) {
        int grow = m0 + wm * 32 + mt * 16 + r + rowadd;
        int gcol = n0 + wn * 64 + nt * 16 + colin;
        if (grow < M && gcol < N) {
          float v = alpha * acc[mt][nt][r];
          if (bias) v += bias[gcol];
          if (act == 1) v = 0.5f * v * (1.0f + erff(v * 0.70710678118f));
          if (cmode == 0) {
            Cf[(long long)grow * ldc + gcol] = v;
          } else if (cmode == 1) {
            Cb[(long long)grow * ldc + gcol] = (__bf16)v;
          } else {  // token row -> (b, d, s)
            int bb = grow >> 9, ss = grow & 511;
            Cvt[(long long)bb * (768ll * 512) + (long long)gcol * 512 + ss] = (__bf16)v;
          }
        }
      }
    }
  }
}

// ---------------------------------------------------------------------------
// f32 -> bf16 linear convert (n multiple of 4).
__global__ __launch_bounds__(256)
void convert_bf16_kernel(const float* __restrict__ in, __bf16* __restrict__ out,
                         long long n) {
  long long i = ((long long)blockIdx.x * 256 + threadIdx.x) * 4;
  if (i + 3 < n) {
    float4 v = *reinterpret_cast<const float4*>(in + i);
    out[i + 0] = (__bf16)v.x; out[i + 1] = (__bf16)v.y;
    out[i + 2] = (__bf16)v.z; out[i + 3] = (__bf16)v.w;
  }
}

// (R x C) f32 -> (C x R) bf16 transpose-convert, 32x32 LDS tiles.
__global__ __launch_bounds__(256)
void transpose_bf16_kernel(const float* __restrict__ in, __bf16* __restrict__ out,
                           int R, int C) {
  __shared__ float tile[32][33];
  int r0 = blockIdx.y * 32, c0 = blockIdx.x * 32;
  int tx = threadIdx.x & 31, ty = threadIdx.x >> 5;
#pragma unroll
  for (int i = 0; i < 32; i += 8) {
    int r = r0 + ty + i, c = c0 + tx;
    tile[ty + i][tx] = (r < R && c < C) ? in[(long long)r * C + c] : 0.f;
  }
  __syncthreads();
#pragma unroll
  for (int i = 0; i < 32; i += 8) {
    int c = c0 + ty + i, r = r0 + tx;
    if (c < C && r < R) out[(long long)c * R + r] = (__bf16)tile[tx][ty + i];
  }
}

// ---------------------------------------------------------------------------
// Row softmax over width 512: f32 scores in, bf16 probs out.
__global__ __launch_bounds__(512)
void softmax512_kernel(const float* __restrict__ s, __bf16* __restrict__ pout) {
  long long row = blockIdx.x;
  const float* p = s + row * 512;
  int tid = threadIdx.x;
  __shared__ float red[512];
  float v = p[tid];
  red[tid] = v;
  __syncthreads();
  for (int st = 256; st > 0; st >>= 1) {
    if (tid < st) red[tid] = fmaxf(red[tid], red[tid + st]);
    __syncthreads();
  }
  float m = red[0];
  __syncthreads();
  float e = __expf(v - m);
  red[tid] = e;
  __syncthreads();
  for (int st = 256; st > 0; st >>= 1) {
    if (tid < st) red[tid] += red[tid + st];
    __syncthreads();
  }
  pout[row * 512 + tid] = (__bf16)(e / red[0]);
}

// ---------------------------------------------------------------------------
// out = LN(a + b(+c)) * g + beta; optional extra bf16 copy of the output.
__global__ __launch_bounds__(256)
void add_ln_kernel(const float* __restrict__ a, const float* __restrict__ b,
                   const float* __restrict__ c, const float* __restrict__ g,
                   const float* __restrict__ beta, float* __restrict__ out,
                   __bf16* __restrict__ outb) {
  long long row = blockIdx.x;
  int tid = threadIdx.x;
  __shared__ float s1[256], s2[256];
  float vals[3];
  float lsum = 0.f, lsq = 0.f;
#pragma unroll
  for (int i = 0; i < 3; i++) {
    int d = tid + i * 256;
    float v = a[row * 768 + d];
    if (b) v += b[row * 768 + d];
    if (c) v += c[row * 768 + d];
    vals[i] = v; lsum += v; lsq += v * v;
  }
  s1[tid] = lsum; s2[tid] = lsq;
  __syncthreads();
  for (int st = 128; st > 0; st >>= 1) {
    if (tid < st) { s1[tid] += s1[tid + st]; s2[tid] += s2[tid + st]; }
    __syncthreads();
  }
  float mean = s1[0] * (1.f / 768.f);
  float var  = s2[0] * (1.f / 768.f) - mean * mean;
  float inv  = rsqrtf(var + 1e-12f);
#pragma unroll
  for (int i = 0; i < 3; i++) {
    int d = tid + i * 256;
    float o = (vals[i] - mean) * inv * g[d] + beta[d];
    if (out)  out[row * 768 + d] = o;
    if (outb) outb[row * 768 + d] = (__bf16)o;
  }
}

// ---------------------------------------------------------------------------
// PKM top-k: per (token, head) top-32 of two 128-score sets, then top-32 of
// the 32x32 cartesian sums; writes softmax weights + value indices.
__global__ __launch_bounds__(128)
void pkm_topk_kernel(const float* __restrict__ s1, const float* __restrict__ s2,
                     float* __restrict__ wout, int* __restrict__ iout) {
  int gid = blockIdx.x * blockDim.x + threadIdx.x;
  if (gid >= 2048 * 4) return;
  int t = gid >> 2;
  int h = gid & 3;
  const float* S1 = s1 + ((long long)h * 2048 + t) * 128;
  const float* S2 = s2 + ((long long)h * 2048 + t) * 128;

  float v1[32], v2[32];
  int   i1[32], i2[32];
  for (int side = 0; side < 2; side++) {
    const float* Sp = side ? S2 : S1;
    float* vv = side ? v2 : v1;
    int*   ii = side ? i2 : i1;
    unsigned long long u0 = 0ull, u1 = 0ull;
    for (int kk = 0; kk < 32; kk++) {
      float best = -1e30f; int bi = 0;
      for (int j = 0; j < 64; j++) {
        float x = Sp[j];
        if (!((u0 >> j) & 1ull) && x > best) { best = x; bi = j; }
      }
      for (int j = 64; j < 128; j++) {
        float x = Sp[j];
        if (!((u1 >> (j - 64)) & 1ull) && x > best) { best = x; bi = j; }
      }
      vv[kk] = best; ii[kk] = bi;
      if (bi < 64) u0 |= 1ull << bi; else u1 |= 1ull << (bi - 64);
    }
  }

  float cs[32]; int ci[32];
  unsigned long long cused[16];
  for (int i = 0; i < 16; i++) cused[i] = 0ull;
  for (int kk = 0; kk < 32; kk++) {
    float best = -1e30f; int bidx = 0;
    for (int a = 0; a < 32; a++) {
      float va = v1[a];
      for (int b2 = 0; b2 < 32; b2++) {
        int idx = a * 32 + b2;
        if (cused[idx >> 6] & (1ull << (idx & 63))) continue;
        float sc = va + v2[b2];
        if (sc > best) { best = sc; bidx = idx; }
      }
    }
    cs[kk] = best; ci[kk] = bidx;
    cused[bidx >> 6] |= 1ull << (bidx & 63);
  }

  float m = cs[0], sum = 0.f;
  for (int kk = 0; kk < 32; kk++) { cs[kk] = __expf(cs[kk] - m); sum += cs[kk]; }
  float invs = 1.f / sum;
  long long ob = (long long)gid * 32;
  for (int kk = 0; kk < 32; kk++) {
    int a = ci[kk] >> 5, b2 = ci[kk] & 31;
    wout[ob + kk] = cs[kk] * invs;
    iout[ob + kk] = i1[a] * 128 + i2[b2];
  }
}

// ---------------------------------------------------------------------------
// PKM gather: out[t,:] = sum_{128 slots} w * values[ind,:].
__global__ __launch_bounds__(256)
void pkm_gather_kernel(const float* __restrict__ w, const int* __restrict__ ind,
                       const float* __restrict__ values, float* __restrict__ out) {
  int t = blockIdx.x;
  int tid = threadIdx.x;
  __shared__ float sw[128];
  __shared__ int   si[128];
  if (tid < 128) {
    sw[tid] = w[(long long)t * 128 + tid];
    si[tid] = ind[(long long)t * 128 + tid];
  }
  __syncthreads();
  float a0 = 0.f, a1 = 0.f, a2 = 0.f;
  int d0 = tid, d1 = tid + 256, d2 = tid + 512;
  for (int i = 0; i < 128; i++) {
    const float* vr = values + (long long)si[i] * 768;
    float wi = sw[i];
    a0 += wi * vr[d0]; a1 += wi * vr[d1]; a2 += wi * vr[d2];
  }
  out[(long long)t * 768 + d0] = a0;
  out[(long long)t * 768 + d1] = a1;
  out[(long long)t * 768 + d2] = a2;
}

// ---------------------------------------------------------------------------
extern "C" void kernel_launch(void* const* d_in, const int* in_sizes, int n_in,
                              void* d_out, int out_size, void* d_ws, size_t ws_size,
                              hipStream_t stream) {
  (void)in_sizes; (void)n_in; (void)out_size; (void)ws_size;
  const float* x    = (const float*)d_in[0];
  const float* Wq   = (const float*)d_in[1];
  const float* bq   = (const float*)d_in[2];
  const float* Wk   = (const float*)d_in[3];
  const float* bk   = (const float*)d_in[4];
  const float* Wv   = (const float*)d_in[5];
  const float* bv   = (const float*)d_in[6];
  const float* Wo   = (const float*)d_in[7];
  const float* bo   = (const float*)d_in[8];
  const float* g1   = (const float*)d_in[9];
  const float* b1   = (const float*)d_in[10];
  const float* Wi   = (const float*)d_in[11];
  const float* bi   = (const float*)d_in[12];
  const float* Wd   = (const float*)d_in[13];
  const float* bd   = (const float*)d_in[14];
  const float* g2   = (const float*)d_in[15];
  const float* b2   = (const float*)d_in[16];
  const float* Wpq  = (const float*)d_in[17];
  const float* bpq  = (const float*)d_in[18];
  const float* keys = (const float*)d_in[19];
  const float* vals = (const float*)d_in[20];
  float* out = (float*)d_out;

  char* wsb = (char*)d_ws;
  size_t off = 0;
  auto takeb = [&](size_t bytes) -> void* {
    void* p = wsb + off;
    off += (bytes + 255) & ~(size_t)255;
    return p;
  };
  // bf16 buffers
  __bf16* xb    = (__bf16*)takeb(2048ull * 768 * 2);
  __bf16* Wqt   = (__bf16*)takeb(768ull * 768 * 2);
  __bf16* Wkt   = (__bf16*)takeb(768ull * 768 * 2);
  __bf16* Wvt   = (__bf16*)takeb(768ull * 768 * 2);
  __bf16* Wot   = (__bf16*)takeb(768ull * 768 * 2);
  __bf16* Wit   = (__bf16*)takeb(3072ull * 768 * 2);
  __bf16* Wdt   = (__bf16*)takeb(768ull * 3072 * 2);
  __bf16* Wpqt  = (__bf16*)takeb(2048ull * 3072 * 2);
  __bf16* keyb  = (__bf16*)takeb(4ull * 2 * 128 * 256 * 2);
  __bf16* qb    = (__bf16*)takeb(2048ull * 768 * 2);
  __bf16* kb    = (__bf16*)takeb(2048ull * 768 * 2);
  __bf16* vt    = (__bf16*)takeb(4ull * 768 * 512 * 2);   // (B, D, S)
  __bf16* prb   = (__bf16*)takeb(48ull * 512 * 512 * 2);  // probs
  __bf16* ctxb  = (__bf16*)takeb(2048ull * 768 * 2);
  __bf16* attnb = (__bf16*)takeb(2048ull * 768 * 2);
  __bf16* intb  = (__bf16*)takeb(2048ull * 3072 * 2);
  __bf16* pqb   = (__bf16*)takeb(2048ull * 2048 * 2);
  // f32 buffers
  float* sc   = (float*)takeb(48ull * 512 * 512 * 4);
  float* proj = (float*)takeb(2048ull * 768 * 4);
  float* attn = (float*)takeb(2048ull * 768 * 4);
  float* dens = (float*)takeb(2048ull * 768 * 4);
  float* s1b  = (float*)takeb(4ull * 2048 * 128 * 4);
  float* s2b  = (float*)takeb(4ull * 2048 * 128 * 4);
  float* wbuf = (float*)takeb(2048ull * 4 * 32 * 4);
  int*   ibuf = (int*)takeb(2048ull * 4 * 32 * 4);
  float* pkmo = (float*)takeb(2048ull * 768 * 4);

  const dim3 T256(256);
  const long long SD = 512ll * 768;
  const long long SS = 512ll * 512;

  // --- one-time bf16 conversions / weight transposes ---
  convert_bf16_kernel<<<dim3(1536), T256, 0, stream>>>(x, xb, 2048ll * 768);
  transpose_bf16_kernel<<<dim3(24, 24), T256, 0, stream>>>(Wq, Wqt, 768, 768);
  transpose_bf16_kernel<<<dim3(24, 24), T256, 0, stream>>>(Wk, Wkt, 768, 768);
  transpose_bf16_kernel<<<dim3(24, 24), T256, 0, stream>>>(Wv, Wvt, 768, 768);
  transpose_bf16_kernel<<<dim3(24, 24), T256, 0, stream>>>(Wo, Wot, 768, 768);
  transpose_bf16_kernel<<<dim3(96, 24), T256, 0, stream>>>(Wi, Wit, 768, 3072);
  transpose_bf16_kernel<<<dim3(24, 96), T256, 0, stream>>>(Wd, Wdt, 3072, 768);
  transpose_bf16_kernel<<<dim3(64, 96), T256, 0, stream>>>(Wpq, Wpqt, 3072, 2048);
  convert_bf16_kernel<<<dim3(256), T256, 0, stream>>>(keys, keyb, 4ll * 2 * 128 * 256);

  // --- QKV projections (V stored transposed per head) ---
  gemm_bf16_wmma_kernel<<<dim3(6, 16, 1), T256, 0, stream>>>(
      xb, 768, 0, 0, Wqt, 768, 0, 0, qb, 768, 0, 0, bq, 2048, 768, 768, 1.f, 0, 1, 1);
  gemm_bf16_wmma_kernel<<<dim3(6, 16, 1), T256, 0, stream>>>(
      xb, 768, 0, 0, Wkt, 768, 0, 0, kb, 768, 0, 0, bk, 2048, 768, 768, 1.f, 0, 1, 1);
  gemm_bf16_wmma_kernel<<<dim3(6, 16, 1), T256, 0, stream>>>(
      xb, 768, 0, 0, Wvt, 768, 0, 0, vt, 768, 0, 0, bv, 2048, 768, 768, 1.f, 0, 1, 2);

  // --- scores[b,h] = q @ k^T / 8 ---
  gemm_bf16_wmma_kernel<<<dim3(4, 4, 48), T256, 0, stream>>>(
      qb, 768, SD, 64, kb, 768, SD, 64,
      sc, 512, 12 * SS, SS, nullptr, 512, 512, 64, 0.125f, 0, 12, 0);

  softmax512_kernel<<<dim3(48 * 512), dim3(512), 0, stream>>>(sc, prb);

  // --- ctx[b,h] = probs @ v  (B operand is vt in (B,D,S)) ---
  gemm_bf16_wmma_kernel<<<dim3(1, 4, 48), T256, 0, stream>>>(
      prb, 512, 12 * SS, SS, vt, 512, 768ll * 512, 64ll * 512,
      ctxb, 768, SD, 64, nullptr, 512, 64, 512, 1.f, 0, 12, 1);

  // --- output projection + LN1 ---
  gemm_bf16_wmma_kernel<<<dim3(6, 16, 1), T256, 0, stream>>>(
      ctxb, 768, 0, 0, Wot, 768, 0, 0, proj, 768, 0, 0, bo, 2048, 768, 768, 1.f, 0, 1, 0);
  add_ln_kernel<<<dim3(2048), T256, 0, stream>>>(proj, x, nullptr, g1, b1, attn, attnb);

  // --- FFN ---
  gemm_bf16_wmma_kernel<<<dim3(24, 16, 1), T256, 0, stream>>>(
      attnb, 768, 0, 0, Wit, 768, 0, 0, intb, 3072, 0, 0, bi, 2048, 3072, 768, 1.f, 1, 1, 1);
  gemm_bf16_wmma_kernel<<<dim3(6, 16, 1), T256, 0, stream>>>(
      intb, 3072, 0, 0, Wdt, 3072, 0, 0, dens, 768, 0, 0, bd, 2048, 768, 3072, 1.f, 0, 1, 0);

  // --- PKM queries + key scores ---
  gemm_bf16_wmma_kernel<<<dim3(16, 16, 1), T256, 0, stream>>>(
      intb, 3072, 0, 0, Wpqt, 3072, 0, 0, pqb, 2048, 0, 0, bpq, 2048, 2048, 3072, 1.f, 0, 1, 1);
  gemm_bf16_wmma_kernel<<<dim3(1, 16, 4), T256, 0, stream>>>(
      pqb, 2048, 512, 0, keyb, 256, 2ll * 128 * 256, 0,
      s1b, 128, 2048ll * 128, 0, nullptr, 2048, 128, 256, 1.f, 0, 1, 0);
  gemm_bf16_wmma_kernel<<<dim3(1, 16, 4), T256, 0, stream>>>(
      pqb + 256, 2048, 512, 0, keyb + 128 * 256, 256, 2ll * 128 * 256, 0,
      s2b, 128, 2048ll * 128, 0, nullptr, 2048, 128, 256, 1.f, 0, 1, 0);

  // --- PKM selection + gather ---
  pkm_topk_kernel<<<dim3(64), dim3(128), 0, stream>>>(s1b, s2b, wbuf, ibuf);
  pkm_gather_kernel<<<dim3(2048), T256, 0, stream>>>(wbuf, ibuf, vals, pkmo);

  // --- final LN2 ---
  add_ln_kernel<<<dim3(2048), T256, 0, stream>>>(dens, pkmo, attn, g2, b2, out, nullptr);
}
